// AttentionNetLayer_66056597012669
// MI455X (gfx1250) — matrix operand order repaired and checked by reading, not compiled, not run on hardware
//
#include <hip/hip_runtime.h>
#include <hip/hip_bf16.h>

typedef __attribute__((ext_vector_type(16))) _Float16 v16h;
typedef __attribute__((ext_vector_type(8)))  float    v8f;

#define T_TOK   262144      // 8*1024*32 tokens
#define NPTS    8192
#define BATCH   8
#define SPTS    1024
#define KNBR    32

// ---------------------------------------------------------------- utilities

__device__ __forceinline__ unsigned ballot32(bool p) {
#if __has_builtin(__builtin_amdgcn_ballot_w32)
    return __builtin_amdgcn_ballot_w32(p);
#else
    return (unsigned)__ballot(p);
#endif
}

__device__ __forceinline__ v8f wmma_f16(v16h a, v16h b, v8f c) {
    // D = A(16x32 f16) * B(32x16 f16) + C(16x16 f32)
    return __builtin_amdgcn_wmma_f32_16x16x32_f16(
        /*neg_a=*/false, a, /*neg_b=*/false, b,
        /*c_mod=*/(short)0, c, /*reuse_a=*/false, /*reuse_b=*/false);
}

// Async copy of 16 bytes per lane from global to LDS (gfx1250 ASYNCcnt path).
// lds_dst = LDS byte offset (low 32 bits of generic shared pointer on AMDGPU),
// gsrc    = 64-bit global address.
__device__ __forceinline__ void async_g2lds_b128(unsigned lds_dst, unsigned long long gsrc) {
    asm volatile("global_load_async_to_lds_b128 %0, %1, off"
                 :: "v"(lds_dst), "v"(gsrc)
                 : "memory");
}
__device__ __forceinline__ void wait_asynccnt0() {
    asm volatile("s_wait_asynccnt 0x0" ::: "memory");
}

// A fragment loader: 16-bit A 16x32 layout (contiguous k order):
//   lane L (<16): row M=L,   K = k0 + {0..15}
//   lane L (>=16): row M=L-16, K = k0 + {16..31}
// -> two contiguous 8-half (16B) chunks per lane.
__device__ __forceinline__ v16h load_a_frag(const _Float16* base, int ld,
                                            int m0, int k0, int lane) {
    int m  = m0 + (lane & 15);
    int ko = k0 + ((lane >> 4) << 4);
    const _Float16* p = base + (size_t)m * ld + ko;
    union { v16h v; uint4 u[2]; } t;
    t.u[0] = *(const uint4*)p;
    t.u[1] = *(const uint4*)(p + 8);
    return t.v;
}

// ---------------------------------------------------------------- weight packing
// Pack f32 weight [K x N] row-major into WMMA B-fragment order:
// tiles of 32(K)x16(N); per tile, 32 lanes x 16 halves contiguous.
// B layout: VGPR r, lanes 0-15 rows K=2r,2r+1; lanes 16-31 rows K=2r+16,2r+17;
// column n = n0 + (lane&15). Rows >= K are zero-padded (K padded to Kp).
__global__ void pack_w(const float* __restrict__ src, _Float16* __restrict__ dst,
                       int K, int Kp, int N) {
    int i = blockIdx.x * 256 + threadIdx.x;
    int total = Kp * N;
    if (i >= total) return;
    int half = i & 15;
    int lane = (i >> 4) & 31;
    int tile = i >> 9;
    int ntn  = N >> 4;
    int kb = tile / ntn, nb = tile % ntn;
    int r = half >> 1, hh = half & 1;
    int k = kb * 32 + 2 * r + ((lane >> 4) << 4) + hh;
    int n = nb * 16 + (lane & 15);
    dst[i] = (k < K) ? (_Float16)src[(size_t)k * N + n] : (_Float16)0.0f;
}

// ---------------------------------------------------------------- FPS
__global__ __launch_bounds__(1024)
void fps_kernel(const float* __restrict__ xyz, float* __restrict__ new_xyz) {
    const int b = blockIdx.x;
    const int tid = threadIdx.x;
    const float* x = xyz + (size_t)b * NPTS * 3;
    float dist[8];
#pragma unroll
    for (int j = 0; j < 8; ++j) dist[j] = 1e10f;
    __shared__ float sv[1024];
    __shared__ int   si[1024];
    int far = 0;
    for (int it = 0; it < SPTS; ++it) {
        if (tid == 0) {
            new_xyz[((size_t)b * SPTS + it) * 3 + 0] = x[(size_t)far * 3 + 0];
            new_xyz[((size_t)b * SPTS + it) * 3 + 1] = x[(size_t)far * 3 + 1];
            new_xyz[((size_t)b * SPTS + it) * 3 + 2] = x[(size_t)far * 3 + 2];
        }
        float cx = x[(size_t)far * 3 + 0];
        float cy = x[(size_t)far * 3 + 1];
        float cz = x[(size_t)far * 3 + 2];
        float bestv = -1.0f; int besti = 0x7fffffff;
#pragma unroll
        for (int j = 0; j < 8; ++j) {
            int p = tid + j * 1024;
            float dx = x[(size_t)p * 3 + 0] - cx;
            float dy = x[(size_t)p * 3 + 1] - cy;
            float dz = x[(size_t)p * 3 + 2] - cz;
            float d = dx * dx + dy * dy + dz * dz;
            float nd = fminf(dist[j], d);
            dist[j] = nd;
            if (nd > bestv || (nd == bestv && p < besti)) { bestv = nd; besti = p; }
        }
        sv[tid] = bestv; si[tid] = besti;
        __syncthreads();
        for (int s = 512; s > 0; s >>= 1) {
            if (tid < s) {
                float ov = sv[tid + s]; int oi = si[tid + s];
                if (ov > sv[tid] || (ov == sv[tid] && oi < si[tid])) { sv[tid] = ov; si[tid] = oi; }
            }
            __syncthreads();
        }
        far = si[0];
        __syncthreads();
    }
}

// ---------------------------------------------------------------- ball query
__global__ __launch_bounds__(256)
void ball_kernel(const float* __restrict__ xyz, const float* __restrict__ new_xyz,
                 int* __restrict__ idxws, int* __restrict__ idxout) {
    const int wid  = blockIdx.x * 8 + (threadIdx.x >> 5);
    const int lane = threadIdx.x & 31;
    const int b = wid >> 10;
    const float* x = xyz + (size_t)b * NPTS * 3;
    const float qx = new_xyz[(size_t)wid * 3 + 0];
    const float qy = new_xyz[(size_t)wid * 3 + 1];
    const float qz = new_xyz[(size_t)wid * 3 + 2];
    const float r2 = 0.2f * 0.2f;
    int* dst = idxws + (size_t)wid * KNBR;
    int* out = idxout + (size_t)wid * KNBR;
    int cnt = 0, first = 0;
    for (int base = 0; base < NPTS && cnt < KNBR; base += 32) {
        int p = base + lane;
        float dx = x[(size_t)p * 3 + 0] - qx;
        float dy = x[(size_t)p * 3 + 1] - qy;
        float dz = x[(size_t)p * 3 + 2] - qz;
        bool hit = (dx * dx + dy * dy + dz * dz) < r2;
        unsigned m = ballot32(hit);
        if (m) {
            if (cnt == 0) first = base + __builtin_ctz(m);
            if (hit) {
                int pos = cnt + __builtin_popcount(m & ((1u << lane) - 1u));
                if (pos < KNBR) { dst[pos] = p; out[pos] = p; }
            }
            cnt += __builtin_popcount(m);
            if (cnt > KNBR) cnt = KNBR;
        }
    }
    for (int j = cnt + lane; j < KNBR; j += 32) { dst[j] = first; out[j] = first; }
}

// ---------------------------------------------------------------- gather+concat -> act0 (T x 96 f16)
__global__ void gather_kernel(const float* __restrict__ xyz, const float* __restrict__ pts,
                              const int* __restrict__ idxws, const float* __restrict__ new_xyz,
                              _Float16* __restrict__ act0) {
    size_t i = (size_t)blockIdx.x * 256 + threadIdx.x;
    if (i >= (size_t)T_TOK * 96) return;
    int c = (int)(i % 96);
    size_t t = i / 96;
    int k = (int)(t % KNBR);
    size_t g = t / KNBR;
    int b = (int)(g >> 10);
    float v = 0.0f;
    if (c < 67) {
        int p = idxws[g * KNBR + k];
        if (c < 3) v = xyz[((size_t)b * NPTS + p) * 3 + c] - new_xyz[g * 3 + c];
        else       v = pts[((size_t)b * NPTS + p) * 64 + (c - 3)];
    }
    act0[i] = (_Float16)v;
}

// ---------------------------------------------------------------- generic token GEMM (WMMA f16)
// Y[T x N] = act( X[T x Kp] @ Wp + bias ) (+ Res), all f16 activations, f32 accumulate.
template <bool RELU, bool ADDRES>
__global__ __launch_bounds__(256)
void gemm_tokens(const _Float16* __restrict__ X, const _Float16* __restrict__ Wp,
                 const float* __restrict__ bias, const _Float16* __restrict__ Res,
                 _Float16* __restrict__ Y, int Kp, int Nn) {
    const int lane = threadIdx.x & 31;
    const int wave = threadIdx.x >> 5;
    const int m0 = blockIdx.x * 128 + wave * 16;
    const int n0 = blockIdx.y * 64;
    const int ntn = Nn >> 4;
    v8f acc[4] = {};
    const int nkb = Kp >> 5;
    for (int kb = 0; kb < nkb; ++kb) {
        v16h a = load_a_frag(X, Kp, m0, kb * 32, lane);
        if (kb + 1 < nkb)
            __builtin_prefetch((const void*)(X + (size_t)(m0 + (lane & 15)) * Kp + (kb + 1) * 32), 0, 3);
#pragma unroll
        for (int nt = 0; nt < 4; ++nt) {
            v16h b = *(const v16h*)(Wp + ((size_t)(kb * ntn + (n0 >> 4) + nt) * 32 + lane) * 16);
            acc[nt] = wmma_f16(a, b, acc[nt]);
        }
    }
    const int colb = n0 + (lane & 15);
    const int mb = m0 + ((lane >> 4) << 3);
#pragma unroll
    for (int nt = 0; nt < 4; ++nt) {
        int col = colb + nt * 16;
        float bv = bias[col];
#pragma unroll
        for (int r = 0; r < 8; ++r) {
            float v = acc[nt][r] + bv;
            if (RELU)   v = fmaxf(v, 0.0f);
            if (ADDRES) v += (float)Res[(size_t)(mb + r) * Nn + col];
            Y[(size_t)(mb + r) * Nn + col] = (_Float16)v;
        }
    }
}

// ---------------------------------------------------------------- fused inner attention (per group)
// x:32x64 -> q,k,v:32x320 (WMMA) -> per-head 32x32 softmax(q k^T/8) (WMMA scores)
// -> P@V (WMMA) -> o-proj 320->64 (WMMA) -> Y[T x 64]
__global__ __launch_bounds__(256)
void inner_attn_kernel(const _Float16* __restrict__ X,
                       const _Float16* __restrict__ wq, const float* __restrict__ bq,
                       const _Float16* __restrict__ wk, const float* __restrict__ bk,
                       const _Float16* __restrict__ wv, const float* __restrict__ bv,
                       const _Float16* __restrict__ wo, const float* __restrict__ bo,
                       _Float16* __restrict__ Y) {
    __shared__ __align__(32) _Float16 xs[32 * 64];     //  4 KB
    __shared__ __align__(32) _Float16 qs[32 * 320];    // 20 KB (overlaid by scores/probs later)
    __shared__ __align__(32) _Float16 ks[32 * 320];    // 20 KB (overlaid by attn-out later)
    __shared__ __align__(32) _Float16 vs[32 * 320];    // 20 KB     => 64 KB total
    float*     sc = (float*)qs;   // scores: 5 heads * 32 rows * 32 f32 = 20480 B
    _Float16*  os = ks;           // attention output (32 x 320)
    const int g = blockIdx.x;
    const int tid = threadIdx.x, lane = tid & 31, wave = tid >> 5;

    // async global->LDS copy of the 32x64 x-tile (16B per lane)
    {
        unsigned lds_dst = (unsigned)(size_t)(void*)xs + (unsigned)tid * 16u;
        unsigned long long gsrc =
            (unsigned long long)(const void*)(X + (size_t)g * 2048) + (unsigned long long)tid * 16ull;
        async_g2lds_b128(lds_dst, gsrc);
        wait_asynccnt0();
    }
    __syncthreads();

    // ---- QKV projections: 3 mats * 2 Mtiles * 20 Ntiles = 120 tiles
    for (int t = wave; t < 120; t += 8) {
        int mat = t / 40, rem = t % 40, mt = rem / 20, nt = rem % 20;
        const _Float16* wp = (mat == 0) ? wq : ((mat == 1) ? wk : wv);
        const float*    bp = (mat == 0) ? bq : ((mat == 1) ? bk : bv);
        _Float16*       dp = (mat == 0) ? qs : ((mat == 1) ? ks : vs);
        v8f acc = {};
#pragma unroll
        for (int kb = 0; kb < 2; ++kb) {
            v16h a = load_a_frag(xs, 64, mt * 16, kb * 32, lane);
            v16h b = *(const v16h*)(wp + ((size_t)(kb * 20 + nt) * 32 + lane) * 16);
            acc = wmma_f16(a, b, acc);
        }
        int col = nt * 16 + (lane & 15);
        int mb  = mt * 16 + ((lane >> 4) << 3);
        float bb = bp[col];
#pragma unroll
        for (int r = 0; r < 8; ++r) dp[(size_t)(mb + r) * 320 + col] = (_Float16)(acc[r] + bb);
    }
    __syncthreads();

    // ---- scores = q k^T / 8 : 5 heads * 2 Mtiles * 2 Jtiles = 20 tiles, staged in regs
    v8f accS[3]; int tS[3]; int nS = 0;
    for (int t = wave; t < 20; t += 8) {
        int h = t >> 2, mt = (t >> 1) & 1, jt = t & 1;
        v8f acc = {};
#pragma unroll
        for (int kb = 0; kb < 2; ++kb) {
            v16h a = load_a_frag(qs + h * 64, 320, mt * 16, kb * 32, lane);
            // B[d,j] = k[j][h*64+d]; contiguous in d per lane
            const _Float16* p = ks + (size_t)(jt * 16 + (lane & 15)) * 320
                                   + h * 64 + kb * 32 + ((lane >> 4) << 4);
            v16h b = *(const v16h*)p;
            acc = wmma_f16(a, b, acc);
        }
        accS[nS] = acc; tS[nS] = t; ++nS;
    }
    __syncthreads();               // all reads of qs done -> safe to overlay with sc
    for (int i = 0; i < nS; ++i) {
        int t = tS[i], h = t >> 2, mt = (t >> 1) & 1, jt = t & 1;
        int col = jt * 16 + (lane & 15);
        int mb  = mt * 16 + ((lane >> 4) << 3);
#pragma unroll
        for (int r = 0; r < 8; ++r)
            sc[(size_t)(h * 32 + mb + r) * 32 + col] = accS[i][r] * 0.125f;
    }
    __syncthreads();

    // ---- softmax per (head,row); probs packed f16 in-place at row start
    if (tid < 160) {
        float* srow = sc + (size_t)tid * 32;
        float v[32];
#pragma unroll
        for (int j = 0; j < 32; ++j) v[j] = srow[j];
        float mx = v[0];
#pragma unroll
        for (int j = 1; j < 32; ++j) mx = fmaxf(mx, v[j]);
        float sum = 0.0f;
#pragma unroll
        for (int j = 0; j < 32; ++j) { float e = __expf(v[j] - mx); v[j] = e; sum += e; }
        float inv = 1.0f / sum;
        _Float16* prow = (_Float16*)srow;
#pragma unroll
        for (int j = 0; j < 32; ++j) prow[j] = (_Float16)(v[j] * inv);
    }
    __syncthreads();

    // ---- O = P @ V : 5 heads * 2 Mtiles * 4 Ntiles = 40 tiles (K=32 -> single WMMA)
    for (int t = wave; t < 40; t += 8) {
        int h = t / 8, rem = t % 8, mt = rem >> 2, nt = rem & 3;
        v8f acc = {};
        v16h a = load_a_frag((const _Float16*)sc + (size_t)h * 2048, 64, mt * 16, 0, lane);
        const _Float16* vb = vs + (size_t)h * 64 + nt * 16 + (lane & 15)
                                + (size_t)((lane >> 4) << 4) * 320;
        v16h b;
#pragma unroll
        for (int r = 0; r < 8; ++r) {
            b[2 * r]     = vb[(size_t)(2 * r) * 320];
            b[2 * r + 1] = vb[(size_t)(2 * r + 1) * 320];
        }
        acc = wmma_f16(a, b, acc);
        int col = h * 64 + nt * 16 + (lane & 15);
        int mb  = mt * 16 + ((lane >> 4) << 3);
#pragma unroll
        for (int r = 0; r < 8; ++r) os[(size_t)(mb + r) * 320 + col] = (_Float16)acc[r];
    }
    __syncthreads();

    // ---- o-projection 320 -> 64: 8 tiles, one per wave, 10 K-steps
    {
        int mt = wave >> 2, nt = wave & 3;
        v8f acc = {};
        for (int kb = 0; kb < 10; ++kb) {
            v16h a = load_a_frag(os, 320, mt * 16, kb * 32, lane);
            v16h b = *(const v16h*)(wo + ((size_t)(kb * 4 + nt) * 32 + lane) * 16);
            acc = wmma_f16(a, b, acc);
        }
        int col = nt * 16 + (lane & 15);
        int mb  = mt * 16 + ((lane >> 4) << 3);
        float bb = bo[col];
#pragma unroll
        for (int r = 0; r < 8; ++r)
            Y[((size_t)g * 32 + mb + r) * 64 + col] = (_Float16)(acc[r] + bb);
    }
}

// ---------------------------------------------------------------- fused final attention (per group)
// q from center token (scalar GEMV, f32 weights); K/V via WMMA in 4-head chunks.
__global__ __launch_bounds__(256)
void final_attn_kernel(const _Float16* __restrict__ X,
                       const float* __restrict__ wqf, const float* __restrict__ bq,
                       const _Float16* __restrict__ wk, const float* __restrict__ bk,
                       const _Float16* __restrict__ wv, const float* __restrict__ bvv,
                       float* __restrict__ Out) {
    __shared__ __align__(32) _Float16 xs[32 * 64];
    __shared__ __align__(32) _Float16 kc[32 * 256];
    __shared__ __align__(32) _Float16 vc[32 * 256];
    __shared__ float qf[1024];
    __shared__ float sc2[4 * 32];
    const int g = blockIdx.x;
    const int tid = threadIdx.x, lane = tid & 31, wave = tid >> 5;

    // async global->LDS copy of the 32x64 x-tile (16B per lane)
    {
        unsigned lds_dst = (unsigned)(size_t)(void*)xs + (unsigned)tid * 16u;
        unsigned long long gsrc =
            (unsigned long long)(const void*)(X + (size_t)g * 2048) + (unsigned long long)tid * 16ull;
        async_g2lds_b128(lds_dst, gsrc);
        wait_asynccnt0();
    }
    __syncthreads();

    // q = x[0] @ Wq + bq (1024 outputs, 4 per thread)
    {
        float acc[4] = {0.f, 0.f, 0.f, 0.f};
        for (int d = 0; d < 64; ++d) {
            float xv = (float)xs[d];
#pragma unroll
            for (int jj = 0; jj < 4; ++jj) acc[jj] += xv * wqf[(size_t)d * 1024 + tid + jj * 256];
        }
#pragma unroll
        for (int jj = 0; jj < 4; ++jj) qf[tid + jj * 256] = acc[jj] + bq[tid + jj * 256];
    }
    __syncthreads();

    for (int hc = 0; hc < 4; ++hc) {     // 4 heads per chunk
        // K,V: 2 mats * 2 Mtiles * 16 Ntiles = 64 tiles
        for (int t = wave; t < 64; t += 8) {
            int mat = t >> 5, rem = t & 31, mt = rem >> 4, nt = rem & 15;
            const _Float16* wp = mat ? wv : wk;
            const float*    bp = mat ? bvv : bk;
            _Float16*       dp = mat ? vc : kc;
            v8f acc = {};
#pragma unroll
            for (int kb = 0; kb < 2; ++kb) {
                v16h a = load_a_frag(xs, 64, mt * 16, kb * 32, lane);
                v16h b = *(const v16h*)(wp + ((size_t)(kb * 64 + hc * 16 + nt) * 32 + lane) * 16);
                acc = wmma_f16(a, b, acc);
            }
            int col = nt * 16 + (lane & 15);
            int mb  = mt * 16 + ((lane >> 4) << 3);
            float bb = bp[hc * 256 + col];
#pragma unroll
            for (int r = 0; r < 8; ++r) dp[(size_t)(mb + r) * 256 + col] = (_Float16)(acc[r] + bb);
        }
        __syncthreads();
        // scores
        if (tid < 128) {
            int hl = tid >> 5, j = tid & 31;
            const float* qh = qf + (hc * 4 + hl) * 64;
            const _Float16* kr = kc + (size_t)j * 256 + hl * 64;
            float s = 0.f;
            for (int d = 0; d < 64; ++d) s += qh[d] * (float)kr[d];
            sc2[hl * 32 + j] = s * 0.125f;
        }
        __syncthreads();
        // softmax (4 rows)
        if (tid < 4) {
            float* row = sc2 + tid * 32;
            float mx = row[0];
            for (int j = 1; j < 32; ++j) mx = fmaxf(mx, row[j]);
            float sum = 0.f;
            for (int j = 0; j < 32; ++j) { float e = __expf(row[j] - mx); row[j] = e; sum += e; }
            float inv = 1.0f / sum;
            for (int j = 0; j < 32; ++j) row[j] *= inv;
        }
        __syncthreads();
        // weighted sum of V
        {
            int hl = tid >> 6, d = tid & 63;
            const float* prow = sc2 + hl * 32;
            const _Float16* vr = vc + hl * 64 + d;
            float o = 0.f;
            for (int j = 0; j < 32; ++j) o += prow[j] * (float)vr[(size_t)j * 256];
            Out[(size_t)g * 1024 + (hc * 4 + hl) * 64 + d] = o;
        }
        __syncthreads();
    }
}

// ---------------------------------------------------------------- host
extern "C" void kernel_launch(void* const* d_in, const int* in_sizes, int n_in,
                              void* d_out, int out_size, void* d_ws, size_t ws_size,
                              hipStream_t stream) {
    (void)in_sizes; (void)n_in; (void)out_size; (void)ws_size;

    const float* xyz = (const float*)d_in[0];
    const float* pts = (const float*)d_in[1];

    // workspace layout
    int*       idxws  = (int*)d_ws;                                          // 1 MB
    _Float16*  wpBase = (_Float16*)((char*)d_ws + (1u << 20));               // 1 MB
    char*      actB   = (char*)d_ws + (2u << 20);
    _Float16*  a0 = (_Float16*)actB;                                         // T x 96
    _Float16*  a1 = (_Float16*)(actB + (size_t)T_TOK * 96 * 2);              // T x 64
    _Float16*  a2 = (_Float16*)(actB + (size_t)T_TOK * (96 + 64) * 2);       // T x 64
    _Float16*  a3 = (_Float16*)(actB + (size_t)T_TOK * (96 + 128) * 2);      // T x 64

    float* outF    = (float*)d_out;
    float* newxyz  = outF;                         // 8*1024*3
    float* outAttn = outF + 24576;                 // 8*1024*1024
    int*   outIdx  = (int*)(outF + 24576 + 8388608);

    // ---- pack weights to WMMA B-fragment layout
    size_t wpo = 0;
    auto packW = [&](int idx, int K, int Kp, int N) -> const _Float16* {
        _Float16* dst = wpBase + wpo;
        int total = Kp * N;
        pack_w<<<(total + 255) / 256, 256, 0, stream>>>((const float*)d_in[idx], dst, K, Kp, N);
        wpo += (size_t)total;
        return dst;
    };
    struct BlkPtrs { const _Float16 *pre[4], *aq, *ak, *av, *ao, *ff[4]; } bp[2];
    for (int blk = 0; blk < 2; ++blk) {
        int base = 2 + blk * 24;
        int din  = (blk == 0) ? 67 : 64;
        int dinp = (blk == 0) ? 96 : 64;
        bp[blk].pre[0] = packW(base + 0, din, dinp, 64);
        bp[blk].pre[1] = packW(base + 2, 64, 64, 64);
        bp[blk].pre[2] = packW(base + 4, 64, 64, 64);
        bp[blk].pre[3] = packW(base + 6, 64, 64, 64);
        bp[blk].aq = packW(base + 8, 64, 64, 320);
        bp[blk].ak = packW(base + 10, 64, 64, 320);
        bp[blk].av = packW(base + 12, 64, 64, 320);
        bp[blk].ao = packW(base + 14, 320, 320, 64);
        for (int l = 0; l < 4; ++l) bp[blk].ff[l] = packW(base + 16 + 2 * l, 64, 64, 64);
    }
    const _Float16* fkW = packW(52, 64, 64, 1024);
    const _Float16* fvW = packW(54, 64, 64, 1024);

    auto fb = [&](int idx) { return (const float*)d_in[idx]; };

    // ---- geometry
    fps_kernel<<<BATCH, 1024, 0, stream>>>(xyz, newxyz);
    ball_kernel<<<1024, 256, 0, stream>>>(xyz, newxyz, idxws, outIdx);
    gather_kernel<<<(unsigned)(((size_t)T_TOK * 96 + 255) / 256), 256, 0, stream>>>(
        xyz, pts, idxws, newxyz, a0);

    // ---- transformer blocks
    const dim3 gGrid(T_TOK / 128, 1);
    const _Float16* cur = a0; int curKp = 96;
    for (int blk = 0; blk < 2; ++blk) {
        int base = 2 + blk * 24;
        gemm_tokens<true,  false><<<gGrid, 256, 0, stream>>>(cur, bp[blk].pre[0], fb(base + 1), nullptr, a1, curKp, 64);
        gemm_tokens<true,  false><<<gGrid, 256, 0, stream>>>(a1,  bp[blk].pre[1], fb(base + 3), nullptr, a2, 64, 64);
        gemm_tokens<true,  false><<<gGrid, 256, 0, stream>>>(a2,  bp[blk].pre[2], fb(base + 5), nullptr, a1, 64, 64);
        gemm_tokens<false, false><<<gGrid, 256, 0, stream>>>(a1,  bp[blk].pre[3], fb(base + 7), nullptr, a2, 64, 64);
        inner_attn_kernel<<<8192, 256, 0, stream>>>(a2,
            bp[blk].aq, fb(base + 9), bp[blk].ak, fb(base + 11),
            bp[blk].av, fb(base + 13), bp[blk].ao, fb(base + 15), a1);
        gemm_tokens<true,  false><<<gGrid, 256, 0, stream>>>(a1, bp[blk].ff[0], fb(base + 17), nullptr, a2, 64, 64);
        gemm_tokens<true,  false><<<gGrid, 256, 0, stream>>>(a2, bp[blk].ff[1], fb(base + 19), nullptr, a3, 64, 64);
        gemm_tokens<true,  false><<<gGrid, 256, 0, stream>>>(a3, bp[blk].ff[2], fb(base + 21), nullptr, a2, 64, 64);
        gemm_tokens<false, true ><<<gGrid, 256, 0, stream>>>(a2, bp[blk].ff[3], fb(base + 23), a1, a3, 64, 64);
        cur = a3; curKp = 64;
    }

    // ---- final attention
    final_attn_kernel<<<8192, 256, 0, stream>>>(a3,
        (const float*)d_in[50], fb(51), fkW, fb(53), fvW, fb(55), outAttn);
}
